// DTSFMEncoder_12704513261599
// MI455X (gfx1250) — compile-verified, matrix operand order checked by reading, not run on hardware
//
#include <hip/hip_runtime.h>

#define N_NODES 50000
#define N_EDGES 1600000
#define IN_DIM  256
#define HID     128
#define LM_DIM  768
#define NEG_SLOPE 0.2f

typedef __attribute__((ext_vector_type(16))) _Float16 v16h;
typedef __attribute__((ext_vector_type(8)))  _Float16 v8h;
typedef __attribute__((ext_vector_type(8)))  float    v8f;
typedef __attribute__((ext_vector_type(4)))  float    v4f;

union V16 { v16h v; v8h h[2]; };

// Monotone float -> uint encoding so we can use integer atomicMax for segment-max.
__device__ __forceinline__ unsigned fenc(float f) {
  unsigned u = __float_as_uint(f);
  return (u & 0x80000000u) ? ~u : (u | 0x80000000u);
}
__device__ __forceinline__ float fdec(unsigned u) {
  return (u & 0x80000000u) ? __uint_as_float(u & 0x7FFFFFFFu) : __uint_as_float(~u);
}

// Convert 8 contiguous f32 -> packed v8h (vector loads, packed converts).
__device__ __forceinline__ v8h cvt8(const float* __restrict__ p) {
  v4f x0 = *(const v4f*)p;
  v4f x1 = *(const v4f*)(p + 4);
  v8h h;
  h[0] = (_Float16)x0[0]; h[1] = (_Float16)x0[1];
  h[2] = (_Float16)x0[2]; h[3] = (_Float16)x0[3];
  h[4] = (_Float16)x1[0]; h[5] = (_Float16)x1[1];
  h[6] = (_Float16)x1[2]; h[7] = (_Float16)x1[3];
  return h;
}

// ---------------------------------------------------------------------------
// Kernel 1: feat = node_feats @ W_gat^T   [50000,256] x [256,128] -> fp32
// f16 WMMA, f32 accumulate. Block = 256 thr = 8 waves; block tile 64 rows x 128 cols.
// Wave (rg,ch): rows rg*16..+15, cols ch*64..+63 (4 x 16x16 WMMA tiles).
// ---------------------------------------------------------------------------
__global__ __launch_bounds__(256)
void k_feat_gemm(const float* __restrict__ X, const float* __restrict__ W,
                 float* __restrict__ feat) {
  __shared__ _Float16 As[64][32];    // A tile (rows x k)
  __shared__ _Float16 Bs[128][32];   // B tile: Bs[n][k] = W[n][kb*32+k]
  const int t      = threadIdx.x;
  const int lane   = t & 31;
  const int wave   = t >> 5;         // 0..7
  const int rg     = wave >> 1;      // row group 0..3
  const int ch     = wave & 1;       // col half 0..1
  const int m0     = blockIdx.x * 64;
  const int mlane  = lane & 15;
  const int hiHalf = lane >> 4;      // 0: lanes 0-15, 1: lanes 16-31
  const bool fullBlock = (m0 + 64 <= N_NODES);  // uniform fast path

  // Per-thread staging coordinates (vectorized, 16B LDS stores)
  const int ar = t >> 2, ac = (t & 3) * 8;      // A: 64x32, 8 halfs/thread
  const int bn = t >> 1, bc = (t & 1) * 16;     // B: 128x32, 16 halfs/thread
  const int agr = m0 + ar;

  v8f acc[4];
  for (int i = 0; i < 4; ++i)
    for (int j = 0; j < 8; ++j) acc[i][j] = 0.0f;

  for (int kb = 0; kb < IN_DIM / 32; ++kb) {
    __syncthreads();
    // Stage A: two v4f global loads -> one ds_store_b128 per thread
    {
      v8h pack;
      if (fullBlock || agr < N_NODES) {
        pack = cvt8(X + (long)agr * IN_DIM + kb * 32 + ac);
      } else {
        for (int i = 0; i < 8; ++i) pack[i] = (_Float16)0.0f;
      }
      *(v8h*)(&As[ar][ac]) = pack;
    }
    // Stage B: four v4f global loads -> two ds_store_b128 per thread
    {
      const float* gp = W + (long)bn * IN_DIM + kb * 32 + bc;
      *(v8h*)(&Bs[bn][bc])     = cvt8(gp);
      *(v8h*)(&Bs[bn][bc + 8]) = cvt8(gp + 8);
    }
    __syncthreads();

    // A fragment per ISA 16-bit A 16x32 layout: two 16B chunks from the row.
    V16 a;
    {
      const v8h* row = (const v8h*)(&As[rg * 16 + mlane][0]);
      a.h[0] = row[hiHalf];       // K 0..7  (lanes<16)  / 8..15  (lanes>=16)
      a.h[1] = row[hiHalf + 2];   // K 16..23            / 24..31
    }
    for (int tt = 0; tt < 4; ++tt) {
      V16 b;
      const v8h* row = (const v8h*)(&Bs[ch * 64 + tt * 16 + mlane][0]);
      b.h[0] = row[hiHalf];
      b.h[1] = row[hiHalf + 2];
      acc[tt] = __builtin_amdgcn_wmma_f32_16x16x32_f16(
          false, a.v, false, b.v, (short)0, acc[tt], false, false);
    }
  }

  // C/D layout: VGPR j -> row (hiHalf*8 + j), col = mlane (within 16x16 tile)
  for (int tt = 0; tt < 4; ++tt) {
    int n = ch * 64 + tt * 16 + mlane;
    for (int j = 0; j < 8; ++j) {
      int m = m0 + rg * 16 + hiHalf * 8 + j;
      if (m < N_NODES) feat[(long)m * HID + n] = acc[tt][j];
    }
  }
}

// ---------------------------------------------------------------------------
// Kernel 2: el = feat . attn_l, er = feat . attn_r  (one wave32 per node)
// ---------------------------------------------------------------------------
__global__ __launch_bounds__(256)
void k_el_er(const float* __restrict__ feat, const float* __restrict__ al,
             const float* __restrict__ ar, float* __restrict__ el,
             float* __restrict__ er) {
  int wid  = (int)((blockIdx.x * blockDim.x + threadIdx.x) >> 5);
  int lane = threadIdx.x & 31;
  if (wid >= N_NODES) return;
  v4f f = *(const v4f*)(feat + (long)wid * HID + lane * 4);
  v4f l = *(const v4f*)(al + lane * 4);
  v4f r = *(const v4f*)(ar + lane * 4);
  float sl = f[0]*l[0] + f[1]*l[1] + f[2]*l[2] + f[3]*l[3];
  float sr = f[0]*r[0] + f[1]*r[1] + f[2]*r[2] + f[3]*r[3];
  for (int m = 16; m >= 1; m >>= 1) {
    sl += __shfl_xor(sl, m, 32);
    sr += __shfl_xor(sr, m, 32);
  }
  if (lane == 0) { el[wid] = sl; er[wid] = sr; }
}

// ---------------------------------------------------------------------------
// Kernel 3: segment-max over dst via ordered-uint atomicMax
// ---------------------------------------------------------------------------
__global__ __launch_bounds__(256)
void k_edge_max(const int* __restrict__ src, const int* __restrict__ dst,
                const float* __restrict__ el, const float* __restrict__ er,
                unsigned* __restrict__ emax_enc) {
  int e = blockIdx.x * blockDim.x + threadIdx.x;
  if (e >= N_EDGES) return;
  int d = dst[e];
  float x = el[src[e]] + er[d];
  x = (x >= 0.0f) ? x : x * NEG_SLOPE;
  atomicMax(&emax_enc[d], fenc(x));
}

// ---------------------------------------------------------------------------
// Kernel 4: denom[d] = sum_e exp(e - emax[d])
// ---------------------------------------------------------------------------
__global__ __launch_bounds__(256)
void k_edge_denom(const int* __restrict__ src, const int* __restrict__ dst,
                  const float* __restrict__ el, const float* __restrict__ er,
                  const unsigned* __restrict__ emax_enc, float* __restrict__ denom) {
  int e = blockIdx.x * blockDim.x + threadIdx.x;
  if (e >= N_EDGES) return;
  int d = dst[e];
  float x = el[src[e]] + er[d];
  x = (x >= 0.0f) ? x : x * NEG_SLOPE;
  atomicAdd(&denom[d], __expf(x - fdec(emax_enc[d])));
}

// ---------------------------------------------------------------------------
// Kernel 5: msum[0:128] = sum_e alpha_e * feat[src_e]  (mean pushed inward:
// sum over n of segment_sum == plain sum over edges). One wave per edge slot,
// grid-stride; per-lane float4 accumulators; LDS cross-wave reduce; 128
// atomics per block only. feat (25.6 MB) is L2-resident on the 192 MB L2.
// ---------------------------------------------------------------------------
__global__ __launch_bounds__(256)
void k_edge_msum(const int* __restrict__ src, const int* __restrict__ dst,
                 const float* __restrict__ el, const float* __restrict__ er,
                 const unsigned* __restrict__ emax_enc, const float* __restrict__ denom,
                 const float* __restrict__ feat, float* __restrict__ msum) {
  __shared__ v4f red[8][32];
  int lane = threadIdx.x & 31;
  int wv   = threadIdx.x >> 5;
  int gw   = (int)((blockIdx.x * blockDim.x + threadIdx.x) >> 5);
  int nw   = (int)((gridDim.x * blockDim.x) >> 5);
  v4f acc; acc[0] = acc[1] = acc[2] = acc[3] = 0.0f;
  for (int e = gw; e < N_EDGES; e += nw) {
    int s = src[e], d = dst[e];
    int e2 = e + nw;
    if (e2 < N_EDGES)  // gfx1250 global_prefetch_b8 on next gather row
      __builtin_prefetch(feat + (long)src[e2] * HID + lane * 4, 0, 0);
    float x = el[s] + er[d];
    x = (x >= 0.0f) ? x : x * NEG_SLOPE;
    float alpha = __expf(x - fdec(emax_enc[d])) / denom[d];
    v4f f = *(const v4f*)(feat + (long)s * HID + lane * 4);
    acc[0] += alpha * f[0];
    acc[1] += alpha * f[1];
    acc[2] += alpha * f[2];
    acc[3] += alpha * f[3];
  }
  red[wv][lane] = acc;
  __syncthreads();
  if (wv == 0) {
    v4f s = red[0][lane];
    for (int w = 1; w < 8; ++w) {
      s[0] += red[w][lane][0]; s[1] += red[w][lane][1];
      s[2] += red[w][lane][2]; s[3] += red[w][lane][3];
    }
    atomicAdd(&msum[lane * 4 + 0], s[0]);
    atomicAdd(&msum[lane * 4 + 1], s[1]);
    atomicAdd(&msum[lane * 4 + 2], s[2]);
    atomicAdd(&msum[lane * 4 + 3], s[3]);
  }
}

// ---------------------------------------------------------------------------
// Kernel 6: zbar = column-sum of z_q. 192 threads x v4f covers all 768 cols,
// fully coalesced b128 stream (154 MB at HBM rate; dominant HBM traffic).
// ---------------------------------------------------------------------------
__global__ __launch_bounds__(192)
void k_zsum(const float* __restrict__ zq, float* __restrict__ zbar) {
  int t = threadIdx.x;
  v4f a; a[0] = a[1] = a[2] = a[3] = 0.0f;
  for (int r = blockIdx.x; r < N_NODES; r += gridDim.x) {
    v4f x = *(const v4f*)(zq + (long)r * LM_DIM + t * 4);
    a[0] += x[0]; a[1] += x[1]; a[2] += x[2]; a[3] += x[3];
  }
  atomicAdd(&zbar[t * 4 + 0], a[0]);
  atomicAdd(&zbar[t * 4 + 1], a[1]);
  atomicAdd(&zbar[t * 4 + 2], a[2]);
  atomicAdd(&zbar[t * 4 + 3], a[3]);
}

// ---------------------------------------------------------------------------
// Kernel 7: final fusion + projection on the means (1 block, 128 threads)
// out = (ct*(msum/N + b_gat) + cs*((zbar/N) @ Wq^T + bq)) @ Wp^T + bp
// ---------------------------------------------------------------------------
__global__ __launch_bounds__(128)
void k_final(const float* __restrict__ msum, const float* __restrict__ zbar,
             const float* __restrict__ b_gat, const float* __restrict__ Wq,
             const float* __restrict__ bq, const float* __restrict__ w_t,
             const float* __restrict__ w_s, const float* __restrict__ Wp,
             const float* __restrict__ bp, float* __restrict__ out) {
  __shared__ float fused[HID];
  int j = threadIdx.x;
  const float invN = 1.0f / (float)N_NODES;
  float zq_j = 0.0f;
  const float* wq = Wq + (long)j * LM_DIM;
  for (int k = 0; k < LM_DIM; ++k) zq_j += zbar[k] * wq[k];
  zq_j = zq_j * invN + bq[j];
  float ht_j = msum[j] * invN + b_gat[j];
  float et = __expf(w_t[0]), es = __expf(w_s[0]);
  float inv = 1.0f / (et + es);
  float ct = et * inv + 0.1f;
  float cs = es * inv + 0.1f;
  fused[j] = ct * ht_j + cs * zq_j;
  __syncthreads();
  const float* wp = Wp + j * HID;
  float o = bp[j];
  for (int k = 0; k < HID; ++k) o += fused[k] * wp[k];
  out[j] = o;
}

// ---------------------------------------------------------------------------
extern "C" void kernel_launch(void* const* d_in, const int* in_sizes, int n_in,
                              void* d_out, int out_size, void* d_ws, size_t ws_size,
                              hipStream_t stream) {
  const float* node_feats = (const float*)d_in[0];
  const float* z_q        = (const float*)d_in[1];
  const int*   src        = (const int*)d_in[2];
  const int*   dst        = (const int*)d_in[3];
  const float* W_gat      = (const float*)d_in[4];
  const float* attn_l     = (const float*)d_in[5];
  const float* attn_r     = (const float*)d_in[6];
  const float* b_gat      = (const float*)d_in[7];
  const float* Wq         = (const float*)d_in[8];
  const float* bq         = (const float*)d_in[9];
  const float* w_t        = (const float*)d_in[10];
  const float* w_s        = (const float*)d_in[11];
  const float* Wp         = (const float*)d_in[12];
  const float* bp         = (const float*)d_in[13];

  // Workspace layout (floats): feat[N*128] | el[N] | er[N] | emax[N](u32) |
  // denom[N] | zbar[768] | msum[128]   (~26.4 MB total)
  float*    feat  = (float*)d_ws;
  float*    el    = feat + (size_t)N_NODES * HID;
  float*    er    = el + N_NODES;
  unsigned* emax  = (unsigned*)(er + N_NODES);
  float*    denom = (float*)(emax + N_NODES);
  float*    zbar  = denom + N_NODES;
  float*    msum  = zbar + LM_DIM;

  // Zero all accumulators every launch (harness does not re-poison).
  hipMemsetAsync(emax, 0, sizeof(unsigned) * (2 * (size_t)N_NODES + LM_DIM + HID),
                 stream);

  k_feat_gemm<<<(N_NODES + 63) / 64, 256, 0, stream>>>(node_feats, W_gat, feat);
  k_el_er<<<(N_NODES + 7) / 8, 256, 0, stream>>>(feat, attn_l, attn_r, el, er);
  k_edge_max<<<(N_EDGES + 255) / 256, 256, 0, stream>>>(src, dst, el, er, emax);
  k_edge_denom<<<(N_EDGES + 255) / 256, 256, 0, stream>>>(src, dst, el, er, emax, denom);
  k_zsum<<<512, 192, 0, stream>>>(z_q, zbar);
  k_edge_msum<<<2048, 256, 0, stream>>>(src, dst, el, er, emax, denom, feat, msum);
  k_final<<<1, 128, 0, stream>>>(msum, zbar, b_gat, Wq, bq, w_t, w_s, Wp, bp,
                                 (float*)d_out);
}
